// MyLSTM_90881507984051
// MI455X (gfx1250) — matrix operand order repaired and checked by reading, not compiled, not run on hardware
//
#include <hip/hip_runtime.h>
#include <hip/hip_bf16.h>

// ---------------------------------------------------------------------------
// 2-layer LSTM on MI455X (gfx1250, wave32).
//
//  * Weights converted once to bf16, transposed to [N=4096, K=1024] so
//    B-fragments are contiguous 32B loads; 32 MB total -> L2-resident
//    (192 MB L2) across all 512 timesteps: per-step weight reads never
//    touch HBM after the first pass.
//  * Activations (x_t and h) are kept in bf16 storage: they are only ever
//    consumed as WMMA A-operands, so this removes all fp32->bf16 cvt chains
//    from the sequential critical path and halves A-side load bytes.
//    Cell state c stays fp32 (accumulated state, full precision).
//  * Per timestep/layer one fused kernel: G = Xin@Wx + Hprev@Wh + b via
//    v_wmma_f32_16x16x32_bf16, gate nonlinearities + h/c update in the
//    epilogue. Each wave owns an M-PAIR (2 of the 4 M-tiles) at one jt
//    column position and accumulates all 4 gates: each B-fragment feeds
//    2 WMMAs and 16 independent WMMAs per k-step hide L2 latency.
//  * Ping-pong state buffers; 1024 stream-ordered launches (graph-captured
//    once) form the inter-step dependency chain.
// ---------------------------------------------------------------------------

typedef __attribute__((ext_vector_type(16))) __bf16 v16bf_t;
typedef __attribute__((ext_vector_type(8)))  __bf16 v8bf_t;
typedef __attribute__((ext_vector_type(4)))  __bf16 v4bf_t;
typedef __attribute__((ext_vector_type(8)))  float  v8f_t;

#define H_DIM 1024
#define G_DIM 4096   // 4 * H_DIM
#define B_DIM 64
#define S_LEN 512

__device__ __forceinline__ v8f_t wmma_bf16(v16bf_t a, v16bf_t b, v8f_t c) {
    // 8 args: (neg_a, A, neg_b, B, c_mod, C, reuse_a, reuse_b)
    return __builtin_amdgcn_wmma_f32_16x16x32_bf16(
        false, a, false, b, (short)0, c, false, false);
}

// A-fragment: 16x32 bf16 tile of row-major bf16 matrix X (ld = 1024).
// Contiguous k order (same as load_b_frag): lanes 0-15 : M=lane,    K groups {kk+0..7,   kk+8..15}
//             lanes 16-31: M=lane-16, K groups {kk+16..23, kk+24..31}
// Each K-group of 8 bf16 = 16 contiguous bytes = one b128 load.
__device__ __forceinline__ v16bf_t load_a_frag(const __bf16* __restrict__ X,
                                               int row, int kk, int lane) {
    const int base0 = kk + ((lane & 16) ? 16 : 0);
    v8bf_t lo = *(const v8bf_t*)(X + (size_t)row * H_DIM + base0);
    v8bf_t hi = *(const v8bf_t*)(X + (size_t)row * H_DIM + base0 + 8);
    return __builtin_shufflevector(lo, hi, 0, 1, 2, 3, 4, 5, 6, 7,
                                           8, 9, 10, 11, 12, 13, 14, 15);
}

// B-fragment: 32x16 bf16 tile of Wt stored [N, K] (contiguous along K).
//   lanes 0-15 : N=n,      K = kk+0 .. kk+15
//   lanes 16-31: N=n-16,   K = kk+16 .. kk+31
// 16 consecutive bf16 = 32 bytes (two b128 loads), 32B-aligned.
__device__ __forceinline__ v16bf_t load_b_frag(const __bf16* __restrict__ Wt,
                                               int n, int kk, int lane) {
    const int kb = kk + ((lane & 16) ? 16 : 0);
    return *(const v16bf_t*)(Wt + (size_t)n * H_DIM + kb);
}

__device__ __forceinline__ float sigmoidf_fast(float x) {
    return 1.0f / (1.0f + __expf(-x));
}

// One LSTM layer, one timestep.
// Grid: 32 blocks x 128 threads (4 waves) = 128 waves = 2 m-pairs x 64 jt.
__global__ void __launch_bounds__(128)
lstm_step_kernel(const __bf16* __restrict__ Xin,    // [64,1024] bf16 layer input
                 const __bf16* __restrict__ Hprev,  // [64,1024] bf16
                 const float*  __restrict__ Cprev,  // [64,1024] fp32
                 const __bf16* __restrict__ WxT,    // [4096,1024] bf16
                 const __bf16* __restrict__ WhT,    // [4096,1024] bf16
                 const float*  __restrict__ bias,   // [4096]
                 __bf16*       __restrict__ Hout,   // [64,1024] bf16
                 float*        __restrict__ Cout)   // [64,1024] fp32
{
    const int wave = blockIdx.x * 4 + (threadIdx.x >> 5); // 0..127
    const int lane = threadIdx.x & 31;
    const int mp = wave >> 6;   // 0..1  : M-pair (rows mp*32 .. mp*32+31)
    const int jt = wave & 63;   // 0..63 : column-within-gate tile

    const int r0   = mp * 32 + (lane & 15);
    const int r1   = r0 + 16;
    const int ncol = jt * 16 + (lane & 15);

    v8f_t acc[2][4] = {};  // [m-tile][gate]

    for (int kk = 0; kk < H_DIM; kk += 32) {
        // ---- input path: issue all loads, then 8 independent WMMAs ----
        v16bf_t ax0 = load_a_frag(Xin, r0, kk, lane);
        v16bf_t ax1 = load_a_frag(Xin, r1, kk, lane);
        v16bf_t bx[4];
#pragma unroll
        for (int g = 0; g < 4; ++g)
            bx[g] = load_b_frag(WxT, g * H_DIM + ncol, kk, lane);
        // Prefetch next k-step's weight cachelines (global_prefetch).
#pragma unroll
        for (int g = 0; g < 4; ++g) {
            __builtin_prefetch(WxT + (size_t)(g * H_DIM + ncol) * H_DIM + kk + 32, 0, 3);
            __builtin_prefetch(WhT + (size_t)(g * H_DIM + ncol) * H_DIM + kk + 32, 0, 3);
        }
#pragma unroll
        for (int g = 0; g < 4; ++g) {
            acc[0][g] = wmma_bf16(ax0, bx[g], acc[0][g]);
            acc[1][g] = wmma_bf16(ax1, bx[g], acc[1][g]);
        }
        // ---- recurrent path ----
        v16bf_t ah0 = load_a_frag(Hprev, r0, kk, lane);
        v16bf_t ah1 = load_a_frag(Hprev, r1, kk, lane);
        v16bf_t bh[4];
#pragma unroll
        for (int g = 0; g < 4; ++g)
            bh[g] = load_b_frag(WhT, g * H_DIM + ncol, kk, lane);
#pragma unroll
        for (int g = 0; g < 4; ++g) {
            acc[0][g] = wmma_bf16(ah0, bh[g], acc[0][g]);
            acc[1][g] = wmma_bf16(ah1, bh[g], acc[1][g]);
        }
    }

    // Epilogue: bias + gates + state update.
    // C/D layout: VGPR v = element (row = v + (lane<16?0:8), col = lane&15).
    const float bi = bias[0 * H_DIM + ncol];
    const float bf = bias[1 * H_DIM + ncol];
    const float bo = bias[2 * H_DIM + ncol];
    const float bc = bias[3 * H_DIM + ncol];

#pragma unroll
    for (int m = 0; m < 2; ++m) {
        const int rbase = mp * 32 + m * 16 + ((lane & 16) ? 8 : 0);
#pragma unroll
        for (int v = 0; v < 8; ++v) {
            const int r = rbase + v;
            const float gi = sigmoidf_fast(acc[m][0][v] + bi);
            const float gf = sigmoidf_fast(acc[m][1][v] + bf);
            const float go = sigmoidf_fast(acc[m][2][v] + bo);
            const float gc = tanhf(acc[m][3][v] + bc);
            const float cold = Cprev[r * H_DIM + ncol];
            // NOTE: h_new uses the OLD cell state (faithful to the reference).
            Hout[r * H_DIM + ncol] = (__bf16)(go * tanhf(cold));
            Cout[r * H_DIM + ncol] = gf * cold + gi * gc;
        }
    }
}

// fp32 [K=1024, N=4096] row-major -> bf16 [N=4096, K=1024] row-major.
// LDS-tiled 32x32 transpose, coalesced on both sides.
__global__ void __launch_bounds__(256)
weight_transpose_kernel(const float* __restrict__ W, __bf16* __restrict__ Wt)
{
    __shared__ __bf16 tile[32][33];
    const int bn = blockIdx.x * 32;
    const int bk = blockIdx.y * 32;
    const int tx = threadIdx.x;     // 0..31
    const int ty = threadIdx.y;     // 0..7
#pragma unroll
    for (int i = 0; i < 32; i += 8)
        tile[ty + i][tx] = (__bf16)W[(size_t)(bk + ty + i) * G_DIM + (bn + tx)];
    __syncthreads();
#pragma unroll
    for (int i = 0; i < 32; i += 8)
        Wt[(size_t)(bn + ty + i) * H_DIM + (bk + tx)] = tile[tx][ty + i];
}

// Bulk fp32 -> bf16 (4 elements/thread, vectorized load+store).
__global__ void __launch_bounds__(256)
f32_to_bf16_kernel(const float* __restrict__ in, __bf16* __restrict__ out, int n4)
{
    const int i = blockIdx.x * blockDim.x + threadIdx.x;
    if (i < n4) {
        float4 v = ((const float4*)in)[i];
        v4bf_t o;
        o[0] = (__bf16)v.x; o[1] = (__bf16)v.y;
        o[2] = (__bf16)v.z; o[3] = (__bf16)v.w;
        ((v4bf_t*)out)[i] = o;
    }
}

// Final bf16 hidden state -> fp32 output.
__global__ void __launch_bounds__(256)
bf16_to_f32_kernel(const __bf16* __restrict__ in, float* __restrict__ out, int n)
{
    const int i = blockIdx.x * blockDim.x + threadIdx.x;
    if (i < n) out[i] = (float)in[i];
}

extern "C" void kernel_launch(void* const* d_in, const int* in_sizes, int n_in,
                              void* d_out, int out_size, void* d_ws, size_t ws_size,
                              hipStream_t stream) {
    (void)in_sizes; (void)n_in; (void)out_size; (void)ws_size;

    const float* inputs = (const float*)d_in[0];  // [512, 64, 1024]
    const float* Wx0    = (const float*)d_in[1];  // [1024, 4096]
    const float* Wh0    = (const float*)d_in[2];  // [1024, 4096]
    const float* b0     = (const float*)d_in[3];  // [4096]
    const float* Wx     = (const float*)d_in[4];  // [1024, 4096]
    const float* Wh     = (const float*)d_in[5];  // [1024, 4096]
    const float* b      = (const float*)d_in[6];  // [4096]

    // Workspace layout (all offsets well aligned):
    //   [ 0..32MB)  : 4x bf16 transposed weights (8 MB each)
    //   [32..96MB)  : bf16 copy of the input tensor
    //   [96MB.. )   : bf16 h ping-pong (4 x 128KB), fp32 c ping-pong (4 x 256KB)
    const size_t WB = (size_t)G_DIM * H_DIM * sizeof(__bf16); // 8 MB
    const size_t SN = (size_t)B_DIM * H_DIM;                  // 65536
    char* ws = (char*)d_ws;
    __bf16* Wx0t = (__bf16*)(ws + 0 * WB);
    __bf16* Wh0t = (__bf16*)(ws + 1 * WB);
    __bf16* Wxt  = (__bf16*)(ws + 2 * WB);
    __bf16* Wht  = (__bf16*)(ws + 3 * WB);
    __bf16* Xbf  = (__bf16*)(ws + 4 * WB);                    // S_LEN*SN bf16
    char*  sbase = ws + 4 * WB + (size_t)S_LEN * SN * sizeof(__bf16);
    const size_t HB = SN * sizeof(__bf16);                    // 128 KB
    const size_t CB = SN * sizeof(float);                     // 256 KB
    __bf16* h0[2] = { (__bf16*)(sbase + 0 * HB), (__bf16*)(sbase + 1 * HB) };
    __bf16* h1[2] = { (__bf16*)(sbase + 2 * HB), (__bf16*)(sbase + 3 * HB) };
    char*  cbase  = sbase + 4 * HB;
    float* c0[2]  = { (float*)(cbase + 0 * CB), (float*)(cbase + 1 * CB) };
    float* c1[2]  = { (float*)(cbase + 2 * CB), (float*)(cbase + 3 * CB) };

    // Zero-init all state buffers (deterministic per call; bf16/f32 zero = 0).
    hipMemsetAsync(sbase, 0, 4 * HB + 4 * CB, stream);

    // One-time weight convert+transpose to bf16 [N,K].
    dim3 tgrid(G_DIM / 32, H_DIM / 32), tblk(32, 8);
    weight_transpose_kernel<<<tgrid, tblk, 0, stream>>>(Wx0, Wx0t);
    weight_transpose_kernel<<<tgrid, tblk, 0, stream>>>(Wh0, Wh0t);
    weight_transpose_kernel<<<tgrid, tblk, 0, stream>>>(Wx,  Wxt);
    weight_transpose_kernel<<<tgrid, tblk, 0, stream>>>(Wh,  Wht);

    // One-time input convert fp32 -> bf16.
    const int n4 = (int)((S_LEN * SN) / 4);                   // 8388608
    f32_to_bf16_kernel<<<n4 / 256, 256, 0, stream>>>(inputs, Xbf, n4);

    // Sequential scan over 512 timesteps; stream order = dependency chain.
    int q_final = 0;
    for (int t = 0; t < S_LEN; ++t) {
        const int p = t & 1, q = p ^ 1;
        const __bf16* x_t = Xbf + (size_t)t * SN;
        // Layer 0
        lstm_step_kernel<<<32, 128, 0, stream>>>(
            x_t, h0[p], c0[p], Wx0t, Wh0t, b0, h0[q], c0[q]);
        // Layer 1 (input = freshly written h0[q])
        lstm_step_kernel<<<32, 128, 0, stream>>>(
            h0[q], h1[p], c1[p], Wxt, Wht, b, h1[q], c1[q]);
        q_final = q;
    }

    // Output: final hidden state of layer 1, converted back to fp32.
    bf16_to_f32_kernel<<<(int)(SN / 256), 256, 0, stream>>>(
        h1[q_final], (float*)d_out, (int)SN);
}
